// ConvGRU_24962349924964
// MI455X (gfx1250) — matrix operand order repaired and checked by reading, not compiled, not run on hardware
//
#include <hip/hip_runtime.h>
#include <hip/hip_bf16.h>

// ---------------------------------------------------------------------------
// ConvGRU for MI455X (gfx1250): fused implicit-GEMM ConvGRU step using
// v_wmma_f32_16x16x32_bf16 (bf16 inputs, fp32 accumulate, fp32 gating).
// Dead "r" gate channels (oc 32..63) are never computed.
// Round 3: division-free, compile-time-branched LDS fill; hardware
// v_exp_f32/v_rcp_f32 gating instead of libm tanhf / IEEE divide.
// ---------------------------------------------------------------------------

typedef __attribute__((ext_vector_type(16))) __bf16 v16bf;
typedef __attribute__((ext_vector_type(8)))  float  v8f;
typedef __attribute__((ext_vector_type(4)))  unsigned int v4u;

#define B_SZ   4
#define CIN    16
#define HID    32
#define TT     64
#define HW     64
#define KPAD   448   // 5 ksteps x-region (9 taps *16ch, padded to 10 taps) + 9 ksteps h (9 taps *32ch)
#define KSTEPS 14
#define MROWS  64    // packed oc: 0..31 -> z(oc 0..31), 32..63 -> n(oc 64..95)
#define CHP    56    // padded channel slots per (row,col): 0..15 x, 16..47 h, 48..55 pad
#define NCOL   66    // halo columns: col = xx+1, xx in [-1,64]

__device__ __forceinline__ unsigned short f32_to_bf16(float f) {
    union { float f; unsigned u; } x; x.f = f;
    unsigned u = x.u;
    unsigned r = u + 0x7fffu + ((u >> 16) & 1u);   // round-to-nearest-even
    return (unsigned short)(r >> 16);
}

__device__ __forceinline__ float fast_sigmoid(float v) {
    return __builtin_amdgcn_rcpf(1.0f + __expf(-v));
}
__device__ __forceinline__ float fast_tanh(float v) {
    // tanh(x) = 1 - 2/(1+e^{2x}); exact saturation at +/-inf
    return 1.0f - 2.0f * __builtin_amdgcn_rcpf(1.0f + __expf(2.0f * v));
}

// K semantic (shared by A-prep and B-fragment addressing):
//   ks = k>>5, kl = k&31
//   ks < 5 : tap p = 2*ks + (kl>>4), ch c = kl&15 ; value = (p<9) ? Wx[oc][c][p] : 0
//   ks >= 5: tap q = ks-5,           ch cin = kl  ; value = Wh[oc][cin][q]
__global__ void convgru_prep(const float* __restrict__ Wx,
                             const float* __restrict__ bx,
                             const float* __restrict__ Wh,
                             unsigned short* __restrict__ Wc,
                             float* __restrict__ bc) {
    int idx = blockIdx.x * blockDim.x + threadIdx.x;
    if (idx < MROWS) {
        int oc = (idx < HID) ? idx : (idx + HID);
        bc[idx] = bx[oc];
    }
    if (idx >= MROWS * KPAD) return;
    int m  = idx / KPAD;
    int k  = idx - m * KPAD;
    int oc = (m < HID) ? m : (m + HID);
    int ks = k >> 5, kl = k & 31;
    float v = 0.0f;
    if (ks < 5) {
        int p = ks * 2 + (kl >> 4);
        int c = kl & 15;
        if (p < 9) v = Wx[oc * 144 + c * 9 + p];
    } else {
        int q = ks - 5;
        v = Wh[oc * 288 + kl * 9 + q];
    }
    Wc[m * KPAD + k] = f32_to_bf16(v);
}

__global__ void convgru_zero(float* __restrict__ p, int n) {
    int i = blockIdx.x * blockDim.x + threadIdx.x;
    if (i < n) p[i] = 0.0f;
}

// One GRU step. grid = (64 rows, 4 batch), block = 128 threads (4 waves).
__global__ __launch_bounds__(128) void convgru_step(
    const float* __restrict__ x,          // (4,16,64,64,64)
    const unsigned short* __restrict__ Wc,// bf16 [64][448]
    const float* __restrict__ bc,         // [64]
    const float* __restrict__ hsrc,       // (4,32,64,64)
    float* __restrict__ hdst,             // (4,32,64,64)
    int t) {
    const int y   = blockIdx.x;
    const int b   = blockIdx.y;
    const int tid = threadIdx.x;

    // RawB[(row*66 + col)*CHP + slot], bf16. col = xx+1. 22176 bytes.
    __shared__ unsigned short RawB[3 * NCOL * CHP];

    // ---- Phase 1a: zero halo border columns (col 0 and 65), all slots. Div-free.
    {
        int slot = tid & 63;
        int col  = (tid >> 6) * (NCOL - 1);   // 0 or 65
        if (slot < CHP) {
            #pragma unroll
            for (int row = 0; row < 3; ++row)
                RawB[(row * NCOL + col) * CHP + slot] = 0;
        }
    }
    // ---- Phase 1b: coalesced float4 fill, x -> slots 0..15, h -> 16..47.
    // tid = ccbase*16 + seg; cc = ccbase + 8*j (j<2: x, j>=2: h, compile-time).
    {
        const int seg    = tid & 15;
        const int ccbase = tid >> 4;          // 0..7
        unsigned short* dcol = &RawB[(seg * 4 + 1) * CHP + ccbase];
        #pragma unroll
        for (int row = 0; row < 3; ++row) {
            const int yy = y + row - 1;
            const bool vrow = (unsigned)yy < (unsigned)HW;
            #pragma unroll
            for (int j = 0; j < 6; ++j) {
                const int cc = ccbase + 8 * j;
                float4 val = make_float4(0.f, 0.f, 0.f, 0.f);
                if (vrow) {
                    if (j < 2)
                        val = *(const float4*)(x + ((((b * CIN + cc) * TT + t) * HW + yy) * HW + seg * 4));
                    else
                        val = *(const float4*)(hsrc + (((b * HID + (cc - CIN)) * HW + yy) * HW + seg * 4));
                }
                unsigned short* dst = dcol + row * (NCOL * CHP) + 8 * j;
                dst[0 * CHP] = f32_to_bf16(val.x);
                dst[1 * CHP] = f32_to_bf16(val.y);
                dst[2 * CHP] = f32_to_bf16(val.z);
                dst[3 * CHP] = f32_to_bf16(val.w);
            }
        }
    }
    __syncthreads();

    // ---- Phase 2: WMMA GEMM. Wave w owns N-tile [w*16, w*16+16), all 4 M-tiles.
    const int lane = tid & 31;
    const int wave = tid >> 5;
    const int lrow = lane & 15;   // n (B/C) or m (A) index within tile
    const int half = lane >> 4;
    const int n0   = wave * 16;

    union BF { v4u u[2]; v16bf b; };

    v8f acc[4];
    #pragma unroll
    for (int mt = 0; mt < 4; ++mt) {
        v8f a;
        #pragma unroll
        for (int v = 0; v < 8; ++v) a[v] = bc[mt * 16 + v + 8 * half];
        acc[mt] = a;
    }

    // x-region ksteps: tap p = 2*ks + half (p==9 -> weights are zero; clamp addr)
    #pragma unroll
    for (int ks = 0; ks < 5; ++ks) {
        int p  = ks * 2 + half;
        int pc = (p > 8) ? 8 : p;
        int dy = (pc >= 6) ? 2 : ((pc >= 3) ? 1 : 0);
        int dx = pc - dy * 3;
        const unsigned short* bp = &RawB[((dy * NCOL) + n0 + lrow + dx) * CHP];  // slots 0..15
        BF bf;
        bf.u[0] = *(const v4u*)(bp);
        bf.u[1] = *(const v4u*)(bp + 8);
        #pragma unroll
        for (int mt = 0; mt < 4; ++mt) {
            BF af;
            const unsigned short* ap = &Wc[(mt * 16 + lrow) * KPAD + ks * 32 + half * 16];
            af.u[0] = *(const v4u*)(ap);
            af.u[1] = *(const v4u*)(ap + 8);
            acc[mt] = __builtin_amdgcn_wmma_f32_16x16x32_bf16(
                false, af.b, false, bf.b, (short)0, acc[mt], false, false);
        }
    }
    // h-region ksteps: tap q = ks-5 (dy,dx compile-time), slots 16 + half*16 + [0..15]
    #pragma unroll
    for (int ks = 5; ks < KSTEPS; ++ks) {
        const int q  = ks - 5;
        const int dy = q / 3;
        const int dx = q - dy * 3;
        const unsigned short* bp =
            &RawB[((dy * NCOL) + n0 + lrow + dx) * CHP + 16 + half * 16];
        BF bf;
        bf.u[0] = *(const v4u*)(bp);
        bf.u[1] = *(const v4u*)(bp + 8);
        #pragma unroll
        for (int mt = 0; mt < 4; ++mt) {
            BF af;
            const unsigned short* ap = &Wc[(mt * 16 + lrow) * KPAD + ks * 32 + half * 16];
            af.u[0] = *(const v4u*)(ap);
            af.u[1] = *(const v4u*)(ap + 8);
            acc[mt] = __builtin_amdgcn_wmma_f32_16x16x32_bf16(
                false, af.b, false, bf.b, (short)0, acc[mt], false, false);
        }
    }

    // ---- Phase 3: gating with v_exp/v_rcp. C/D layout: N = lane&15, M = v + 8*half.
    const int xp   = n0 + lrow;
    const int base = ((b * HID) * HW + y) * HW + xp;
    #pragma unroll
    for (int v = 0; v < 8; ++v) {
        int mz = v + 8 * half;
        float z0 = fast_sigmoid(acc[0][v]);
        float z1 = fast_sigmoid(acc[1][v]);
        float g0 = fast_tanh(acc[2][v]);
        float g1 = fast_tanh(acc[3][v]);
        int i0 = base + mz * (HW * HW);
        int i1 = i0 + 16 * (HW * HW);
        float h0 = hsrc[i0], h1 = hsrc[i1];
        hdst[i0] = (1.0f - z0) * h0 + z0 * g0;
        hdst[i1] = (1.0f - z1) * h1 + z1 * g1;
    }
}

extern "C" void kernel_launch(void* const* d_in, const int* in_sizes, int n_in,
                              void* d_out, int out_size, void* d_ws, size_t ws_size,
                              hipStream_t stream) {
    const float* x  = (const float*)d_in[0];   // (4,16,64,64,64)
    const float* Wx = (const float*)d_in[1];   // (96,16,3,3)
    const float* bx = (const float*)d_in[2];   // (96,)
    const float* Wh = (const float*)d_in[3];   // (96,32,3,3)
    float* out = (float*)d_out;                // (4,32,64,64)

    // Workspace (~4.1 MB): packed bf16 weights + bias + two h ping-pong buffers
    char* ws = (char*)d_ws;
    unsigned short* Wc = (unsigned short*)ws;                    // 64*448*2 = 57344 B
    float* bc = (float*)(ws + 57344);                            // 256 B
    const int HN = B_SZ * HID * HW * HW;                         // 524288 floats
    float* hA = (float*)(ws + 57600);
    float* hB = (float*)(ws + 57600 + (size_t)HN * 4);

    convgru_prep<<<(MROWS * KPAD + 255) / 256, 256, 0, stream>>>(Wx, bx, Wh, Wc, bc);
    convgru_zero<<<(HN + 255) / 256, 256, 0, stream>>>(hA, HN);

    const float* src = hA;
    for (int t = 0; t < TT; ++t) {
        float* dst = (t == TT - 1) ? out : ((t & 1) ? hA : hB);
        convgru_step<<<dim3(HW, B_SZ), 128, 0, stream>>>(x, Wc, bc, src, dst, t);
        src = dst;
    }
}
